// SetAbstractionPair_89438398972561
// MI455X (gfx1250) — matrix operand order repaired and checked by reading, compiled, not run on hardware
//
#include <hip/hip_runtime.h>
#include <hip/hip_bf16.h>

// ---------------------------------------------------------------------------
// PointNet++ SetAbstraction on MI455X (gfx1250, wave32).
//   k1: ball query (radius 0.2, first-32-by-index) -> int nidx[N1][32] in ws
//   k2: repack W1 (67x128 -> pad 96xK) / W2 (128x256) into per-lane bf16
//       WMMA B-fragment order (contiguous 32B per lane per fragment)
//   k3: per-query fused gather + MLP(67->128->256, relu) via
//       v_wmma_f32_16x16x32_bf16 + masked max aggregation -> out[N1][256]
//   Round 2: waves own fixed n-tile strips so B fragments are register-
//   resident across all 3 M-tiles (3x less weight-fragment VMEM traffic).
// ---------------------------------------------------------------------------

typedef __attribute__((ext_vector_type(16))) __bf16          v16bf;
typedef __attribute__((ext_vector_type(16))) unsigned short  v16us;
typedef __attribute__((ext_vector_type(8)))  float           v8f;
typedef __attribute__((ext_vector_type(8)))  unsigned int    v8u;

#define RAD2   0.04f
#define K_NB   32
#define N1P    1024
#define N2P    4096
#define NBATCH 16
#define N1TOT  (NBATCH * N1P)
#define D_IN   64
#define H1     128
#define H2     256
#define KPAD1  96      // 67 padded to 3 k-tiles of 32
#define MROWS  48      // 33 rows (32 nbrs + self) padded to 3 m-tiles of 16
#define W1_KT  3
#define W2_KT  4

#define W1F_ELEMS (W1_KT * 8  * 32 * 16)   // 12288
#define W2F_ELEMS (W2_KT * 16 * 32 * 16)   // 32768

__device__ __forceinline__ unsigned short f2bf(float f) {
  unsigned u = __float_as_uint(f);
  u += 0x7fffu + ((u >> 16) & 1u);        // round-to-nearest-even
  return (unsigned short)(u >> 16);
}

// A fragment (16x32 bf16) from LDS row-major [., ld], ISA 7.12.2 layout:
// lane L: row = L&15 ; element e -> K = (e&7) + 8*( ((e>>3)<<1) | (L>>4) )
__device__ __forceinline__ v16bf load_A_lds(const unsigned short* s, int ld,
                                            int row0, int k0, int lane) {
  const int hf = lane >> 4;
  const int r  = lane & 15;
  const unsigned short* p = s + (row0 + r) * ld + k0;
  v16us t;
#pragma unroll
  for (int e = 0; e < 16; e += 2) {
    const int g  = ((e >> 3) << 1) | hf;
    const int kk = (e & 7) + (g << 3);          // even -> dword aligned
    const unsigned pair = *(const unsigned*)(p + kk);
    t[e]     = (unsigned short)(pair & 0xffffu);
    t[e + 1] = (unsigned short)(pair >> 16);
  }
  return __builtin_bit_cast(v16bf, t);
}

// B fragment (32x16 bf16), pre-swizzled in global: one contiguous 32B per lane
__device__ __forceinline__ v16bf load_B_frag(const unsigned short* w,
                                             int tile, int lane) {
  const v8u raw = *(const v8u*)(w + ((size_t)tile * 32 + lane) * 16);
  return __builtin_bit_cast(v16bf, raw);
}

// ------------------------------- kernel 1 ----------------------------------
__global__ void __launch_bounds__(256)
ballq_kernel(const float* __restrict__ pos1, const float* __restrict__ pos2,
             int* __restrict__ nidx) {
  const int lane = threadIdx.x & 31;
  const int q    = blockIdx.x * 8 + (threadIdx.x >> 5);
  const int base2 = (q >> 10) << 12;              // batch * N2P
  const float qx = pos1[q * 3 + 0];
  const float qy = pos1[q * 3 + 1];
  const float qz = pos1[q * 3 + 2];
  nidx[q * K_NB + lane] = -1;
  int cnt = 0;
  for (int j0 = 0; j0 < N2P && cnt < K_NB; j0 += 32) {
    const int s = base2 + j0 + lane;
    const float dx = pos2[s * 3 + 0] - qx;
    const float dy = pos2[s * 3 + 1] - qy;
    const float dz = pos2[s * 3 + 2] - qz;
    const bool in = (dx * dx + dy * dy + dz * dz) <= RAD2;
    const unsigned m   = (unsigned)__ballot(in);
    const int      pre = __popc(m & ((1u << lane) - 1u));
    if (in && (cnt + pre) < K_NB) nidx[q * K_NB + cnt + pre] = s;
    cnt += __popc(m);
  }
}

// ------------------------------- kernel 2 ----------------------------------
__global__ void __launch_bounds__(256)
repack_kernel(const float* __restrict__ W1, const float* __restrict__ W2,
              unsigned short* __restrict__ w1f, unsigned short* __restrict__ w2f) {
  const int idx = blockIdx.x * 256 + threadIdx.x;
  if (idx < W1F_ELEMS) {                          // e:4b lane:5b nt:3b kt:2b
    const int e = idx & 15, lane = (idx >> 4) & 31;
    const int nt = (idx >> 9) & 7, kt = idx >> 12;
    const int row = kt * 32 + (e & 7) + 8 * (lane >> 4) + 16 * (e >> 3);
    const int col = nt * 16 + (lane & 15);
    w1f[idx] = f2bf((row < 67) ? W1[row * H1 + col] : 0.f);
  }
  if (idx < W2F_ELEMS) {                          // e:4b lane:5b nt:4b kt:2b
    const int e = idx & 15, lane = (idx >> 4) & 31;
    const int nt = (idx >> 9) & 15, kt = idx >> 13;
    const int row = kt * 32 + (e & 7) + 8 * (lane >> 4) + 16 * (e >> 3);
    const int col = nt * 16 + (lane & 15);
    w2f[idx] = f2bf(W2[row * H2 + col]);
  }
}

// ------------------------------- kernel 3 ----------------------------------
__global__ void __launch_bounds__(256)
mlp_kernel(const float* __restrict__ x2, const float* __restrict__ pos1,
           const float* __restrict__ pos2,
           const unsigned short* __restrict__ w1f, const float* __restrict__ b1,
           const unsigned short* __restrict__ w2f, const float* __restrict__ b2,
           const int* __restrict__ g_nidx, float* __restrict__ out) {
  __shared__ unsigned short s_h[MROWS * KPAD1];   //  9216 B
  __shared__ unsigned short s_m1[MROWS * H1];     // 12288 B
  __shared__ unsigned       s_out[H2];            //  1024 B
  __shared__ int            s_nidx[33];

  const int q    = blockIdx.x;
  const int tid  = threadIdx.x;
  const int lane = tid & 31;
  const int wave = tid >> 5;
  const int hf   = lane >> 4;

  if (tid < 33) s_nidx[tid] = (tid == 32) ? q : g_nidx[q * K_NB + tid];
  s_out[tid] = 0u;                                 // relu >= 0, self valid
  __syncthreads();

  // ---- stage 1: gather h = [x_j | pos_j - pos_i], bf16, zero-padded ----
  for (int idx = tid; idx < MROWS * KPAD1; idx += 256) {
    const int row = idx / KPAD1;
    const int col = idx - row * KPAD1;
    float v = 0.f;
    const int src = (row < 33) ? s_nidx[row] : -1;
    if (src >= 0) {
      if (col < D_IN)       v = x2[src * D_IN + col];
      else if (col < 67) {
        const int d = col - D_IN;
        v = pos2[src * 3 + d] - pos1[q * 3 + d];
      }
    }
    s_h[idx] = f2bf(v);
  }
  __syncthreads();

  // ---- stage 2: layer 1, 48x96 @ 96x128. Wave w owns n-tile w; B frags
  //      stay in registers across all 3 m-tiles. Bias in C, relu->bf16 LDS.
  {
    const int nt  = wave;                          // 8 waves == 8 n-tiles
    const int col = nt * 16 + (lane & 15);
    const float bv = b1[col];
    v16bf bfr[W1_KT];
#pragma unroll
    for (int kt = 0; kt < W1_KT; kt++)
      bfr[kt] = load_B_frag(w1f, kt * 8 + nt, lane);
#pragma unroll
    for (int mt = 0; mt < 3; mt++) {
      v8f acc;
#pragma unroll
      for (int i = 0; i < 8; i++) acc[i] = bv;
#pragma unroll
      for (int kt = 0; kt < W1_KT; kt++) {
        const v16bf a = load_A_lds(s_h, KPAD1, mt * 16, kt * 32, lane);
        acc = __builtin_amdgcn_wmma_f32_16x16x32_bf16(false, a, false, bfr[kt],
                                                      (short)0, acc, false, false);
      }
#pragma unroll
      for (int i = 0; i < 8; i++) {
        const int row = mt * 16 + i + 8 * hf;      // C layout: VGPR i -> M=i+8*hf
        s_m1[row * H1 + col] = f2bf(fmaxf(acc[i], 0.f));
      }
    }
  }
  __syncthreads();

  // ---- stage 3: layer 2, 48x128 @ 128x256. Wave w owns n-tiles {w, w+8};
  //      B frags register-resident across m-tiles; relu + masked max.
  for (int nt = wave; nt < 16; nt += 8) {
    const int col = nt * 16 + (lane & 15);
    const float bv = b2[col];
    v16bf bfr[W2_KT];
#pragma unroll
    for (int kt = 0; kt < W2_KT; kt++)
      bfr[kt] = load_B_frag(w2f, kt * 16 + nt, lane);
#pragma unroll
    for (int mt = 0; mt < 3; mt++) {
      v8f acc;
#pragma unroll
      for (int i = 0; i < 8; i++) acc[i] = bv;
#pragma unroll
      for (int kt = 0; kt < W2_KT; kt++) {
        const v16bf a = load_A_lds(s_m1, H1, mt * 16, kt * 32, lane);
        acc = __builtin_amdgcn_wmma_f32_16x16x32_bf16(false, a, false, bfr[kt],
                                                      (short)0, acc, false, false);
      }
#pragma unroll
      for (int i = 0; i < 8; i++) {
        const int row = mt * 16 + i + 8 * hf;
        if (row < 33 && (row == 32 || s_nidx[row] >= 0)) {
          const float v = fmaxf(acc[i], 0.f);      // nonneg: uint order == float order
          atomicMax(&s_out[col], __float_as_uint(v));
        }
      }
    }
  }
  __syncthreads();

  out[(size_t)q * H2 + tid] = __uint_as_float(s_out[tid]);
}

// ------------------------------- launcher ----------------------------------
extern "C" void kernel_launch(void* const* d_in, const int* in_sizes, int n_in,
                              void* d_out, int out_size, void* d_ws, size_t ws_size,
                              hipStream_t stream) {
  // setup_inputs order: x1, pos1, batch1, x2, pos2, batch2, W1, b1, W2, b2
  const float* pos1 = (const float*)d_in[1];
  const float* x2   = (const float*)d_in[3];
  const float* pos2 = (const float*)d_in[4];
  const float* W1   = (const float*)d_in[6];
  const float* b1   = (const float*)d_in[7];
  const float* W2   = (const float*)d_in[8];
  const float* b2   = (const float*)d_in[9];
  float* out = (float*)d_out;

  int* nidx = (int*)d_ws;                                      // N1*32 ints = 2 MB
  unsigned short* w1f =
      (unsigned short*)((char*)d_ws + (size_t)N1TOT * K_NB * sizeof(int));
  unsigned short* w2f = w1f + W1F_ELEMS;

  ballq_kernel<<<N1TOT / 8, 256, 0, stream>>>(pos1, pos2, nidx);
  repack_kernel<<<W2F_ELEMS / 256, 256, 0, stream>>>(W1, W2, w1f, w2f);
  mlp_kernel<<<N1TOT, 256, 0, stream>>>(x2, pos1, pos2, w1f, b1, w2f, b2,
                                        nidx, out);
}
